// MPNN_29884382446410
// MI455X (gfx1250) — compile-verified, matrix operand order checked by reading, not run on hardware
//
#include <hip/hip_runtime.h>
#include <hip/hip_bf16.h>
#include <math.h>

#define HID 128
#define WPB 4          // waves per block (wave32)
#define K0  384        // concat width for node MLP layer 0

typedef __attribute__((ext_vector_type(16))) __bf16 v16bf;
typedef __attribute__((ext_vector_type(8)))  float  v8f;
typedef __attribute__((ext_vector_type(4)))  unsigned int v4u;

union ABFrag { v16bf v; v4u q[2]; };

__device__ __forceinline__ unsigned short f2bf(float f) {
    unsigned int u = __float_as_uint(f);
    u += 0x7fffu + ((u >> 16) & 1u);      // round-to-nearest-even
    return (unsigned short)(u >> 16);
}

// pack two floats -> two bf16 in one dword (lo in [15:0])
__device__ __forceinline__ unsigned int f2bf2(float lo, float hi) {
    return (unsigned int)f2bf(lo) | ((unsigned int)f2bf(hi) << 16);
}

__device__ __forceinline__ v8f zero8() {
    v8f z;
#pragma unroll
    for (int i = 0; i < 8; ++i) z[i] = 0.f;
    return z;
}

// B-fragment swizzle (32x16 bf16, ISA layout: lane = (n&15)+16*khalf, 16 bf16
// per lane contiguous -> two ds_load_b128 per fragment).
// Packed variant: stores elements (k, k+1), k even. They always share a lane
// (k-half groups are 8-aligned) and occupy adjacent even/odd slots -> one b32.
__device__ __forceinline__ void swz_store32(unsigned short* wbuf, int k, int n,
                                            unsigned int v2) {
    int kc  = k >> 5, kin = k & 31;                 // k even
    int hh  = (kin >> 3) & 1;                       // k-half bit
    int s   = (kin & 7) | ((kin & 16) ? 8 : 0);     // even slot
    unsigned short* p =
        wbuf + ((((kc << 3) + (n >> 4)) << 5) + ((n & 15) + (hh << 4))) * 16 + s;
    *(unsigned int*)p = v2;                         // 4B aligned (s even)
}

// Cooperative (whole block) load of a K x 128 f32 weight slab into swizzled LDS.
__device__ __forceinline__ void load_w(unsigned short* wbuf, const float* W,
                                       int kOff, int kCnt, int tid) {
    for (int e = tid; e < (kCnt >> 1) * HID; e += WPB * 32) {
        int kp = e >> 7, n = e & 127, k = kp * 2;
        unsigned int v2 = f2bf2(W[(size_t)(kOff + k) * HID + n],
                                W[(size_t)(kOff + k + 1) * HID + n]);
        swz_store32(wbuf, k, n, v2);
    }
}

// One wave: acc[nt] += A(16 x kchunks*32, bf16 LDS row-major, stride aStride)
//                    x B(kchunks*32 x 128, swizzled LDS)
__device__ __forceinline__ void gemm_acc(const unsigned short* aBase, int aStride,
                                         const unsigned short* wbuf, int kchunks,
                                         int lane, v8f acc[8]) {
    const int m  = lane & 15;
    const int kb = (lane >> 4) * 8;
#pragma unroll
    for (int kc = 0; kc < kchunks; ++kc) {
        ABFrag a;
        a.q[0] = *(const v4u*)(aBase + m * aStride + kc * 32 + kb);
        a.q[1] = *(const v4u*)(aBase + m * aStride + kc * 32 + 16 + kb);
#pragma unroll
        for (int nt = 0; nt < 8; ++nt) {
            ABFrag b;
            const v4u* bp = (const v4u*)(wbuf + ((kc * 8 + nt) * 32 + lane) * 16);
            b.q[0] = bp[0];
            b.q[1] = bp[1];
            acc[nt] = __builtin_amdgcn_wmma_f32_16x16x32_bf16(
                false, a.v, false, b.v, (short)0, acc[nt], false, false);
        }
    }
}

// bias + ReLU + restage as bf16 A tile (row-major, stride aStride) for next layer
__device__ __forceinline__ void epi_relu_stage(v8f acc[8], const float* bias,
                                               unsigned short* A, int aStride,
                                               int lane) {
    const int nl = lane & 15, hh = lane >> 4;
#pragma unroll
    for (int nt = 0; nt < 8; ++nt) {
        float bv = bias[nt * 16 + nl];
#pragma unroll
        for (int r = 0; r < 8; ++r) {
            float v = acc[nt][r] + bv;
            v = v > 0.f ? v : 0.f;
            A[(r + 8 * hh) * aStride + nt * 16 + nl] = f2bf(v);
        }
    }
}

// ---------------- init MLP: feat[N,16] -> h[N,128] ----------------
__global__ __launch_bounds__(WPB * 32)
void init_mlp_kernel(const float* __restrict__ feat,
                     const float* __restrict__ W0, const float* __restrict__ b0,
                     const float* __restrict__ W1, const float* __restrict__ b1,
                     const float* __restrict__ W2, const float* __restrict__ b2,
                     float* __restrict__ hOut, int nTiles) {
    __shared__ __align__(16) unsigned short wbuf[4 * 8 * 32 * 16];   // 32 KB
    __shared__ __align__(16) unsigned short astage[WPB][16 * HID];   // 16 KB
    const int tid = threadIdx.x, wid = tid >> 5, lane = tid & 31;
    const int tile = blockIdx.x * WPB + wid;
    const bool active = tile < nTiles;
    unsigned short* A = astage[wid];
    const int nl = lane & 15, hh = lane >> 4;

    if (active) {
        // 16x16 feat tile, K padded to 32; packed dword stores
        for (int idx = lane; idx < 16 * 16; idx += 32) {
            int r = idx >> 4, cp = idx & 15, c = cp * 2;
            unsigned int v2 = 0u;
            if (c < 16) {
                float2 fv = *(const float2*)(feat + (size_t)(tile * 16 + r) * 16 + c);
                v2 = f2bf2(fv.x, fv.y);
            }
            *(unsigned int*)(A + r * HID + c) = v2;
        }
    }
    // W0 padded to 32x128, packed k-pairs
    for (int e = tid; e < 16 * HID; e += WPB * 32) {
        int kp = e >> 7, n = e & 127, k = kp * 2;
        unsigned int v2 = 0u;
        if (k < 16)
            v2 = f2bf2(W0[(size_t)k * HID + n], W0[(size_t)(k + 1) * HID + n]);
        swz_store32(wbuf, k, n, v2);
    }
    __syncthreads();

    v8f acc[8];
    if (active) {
#pragma unroll
        for (int nt = 0; nt < 8; ++nt) acc[nt] = zero8();
        gemm_acc(A, HID, wbuf, 1, lane, acc);
        epi_relu_stage(acc, b0, A, HID, lane);
    }
    __syncthreads();
    load_w(wbuf, W1, 0, HID, tid);
    __syncthreads();
    if (active) {
#pragma unroll
        for (int nt = 0; nt < 8; ++nt) acc[nt] = zero8();
        gemm_acc(A, HID, wbuf, 4, lane, acc);
        epi_relu_stage(acc, b1, A, HID, lane);
    }
    __syncthreads();
    load_w(wbuf, W2, 0, HID, tid);
    __syncthreads();
    if (active) {
#pragma unroll
        for (int nt = 0; nt < 8; ++nt) acc[nt] = zero8();
        gemm_acc(A, HID, wbuf, 4, lane, acc);
#pragma unroll
        for (int nt = 0; nt < 8; ++nt) {
            float bv = b2[nt * 16 + nl];
#pragma unroll
            for (int r = 0; r < 8; ++r) {
                size_t g = (size_t)(tile * 16 + r + 8 * hh) * HID + nt * 16 + nl;
                hOut[g] = acc[nt][r] + bv;
            }
        }
    }
}

// ---------------- node MLP: [msg|h|gr][N,384] -> h'[N,128] ----------------
__global__ __launch_bounds__(WPB * 32)
void node_mlp_kernel(const float* __restrict__ msg, const float* __restrict__ hIn,
                     const float* __restrict__ gr, const unsigned int* __restrict__ flags,
                     const float* __restrict__ W0, const float* __restrict__ b0,
                     const float* __restrict__ W1, const float* __restrict__ b1,
                     const float* __restrict__ W2, const float* __restrict__ b2,
                     float* __restrict__ hOut, int nTiles) {
    __shared__ __align__(16) unsigned short wbuf[4 * 8 * 32 * 16];  // 32 KB
    __shared__ __align__(16) unsigned short astage[WPB][16 * K0];   // 48 KB
    const int tid = threadIdx.x, wid = tid >> 5, lane = tid & 31;
    const int tile = blockIdx.x * WPB + wid;
    const bool active = tile < nTiles;
    unsigned short* A = astage[wid];
    const int nl = lane & 15, hh = lane >> 4;

    if (active) {
        // msg -> cols 0..127, h -> 128..255 : packed dword stores, float2 loads
        for (int idx = lane; idx < 16 * 64; idx += 32) {
            int r = idx >> 6, cp = idx & 63, c = cp * 2;
            size_t g = (size_t)(tile * 16 + r) * HID + c;
            float2 mv = *(const float2*)(msg + g);
            float2 hv = *(const float2*)(hIn + g);
            *(unsigned int*)(A + r * K0 + c)       = f2bf2(mv.x, mv.y);
            *(unsigned int*)(A + r * K0 + HID + c) = f2bf2(hv.x, hv.y);
        }
        // gr broadcast -> cols 256..383
        for (int cp = lane; cp < 64; cp += 32) {
            int c = cp * 2;
            unsigned int gv = f2bf2(gr[c], gr[c + 1]);
#pragma unroll
            for (int r = 0; r < 16; ++r)
                *(unsigned int*)(A + r * K0 + 2 * HID + c) = gv;
        }
    }

    v8f acc[8];
#pragma unroll
    for (int nt = 0; nt < 8; ++nt) acc[nt] = zero8();

    // Layer 0: K=384, stream W0 through LDS in three 128-deep slabs
    for (int p = 0; p < 3; ++p) {
        load_w(wbuf, W0, p * HID, HID, tid);
        __syncthreads();
        if (active) gemm_acc(A + p * HID, K0, wbuf, 4, lane, acc);
        __syncthreads();
    }
    if (active) epi_relu_stage(acc, b0, A, K0, lane);

    load_w(wbuf, W1, 0, HID, tid);
    __syncthreads();
    if (active) {
#pragma unroll
        for (int nt = 0; nt < 8; ++nt) acc[nt] = zero8();
        gemm_acc(A, K0, wbuf, 4, lane, acc);
        epi_relu_stage(acc, b1, A, K0, lane);
    }
    __syncthreads();
    load_w(wbuf, W2, 0, HID, tid);
    __syncthreads();
    if (active) {
#pragma unroll
        for (int nt = 0; nt < 8; ++nt) acc[nt] = zero8();
        gemm_acc(A, K0, wbuf, 4, lane, acc);
#pragma unroll
        for (int nt = 0; nt < 8; ++nt) {          // + bias
            float bv = b2[nt * 16 + nl];
#pragma unroll
            for (int r = 0; r < 8; ++r) acc[nt][r] += bv;
        }
        // row L2 norms: lane holds 8 cols per row; 16-lane column groups cover
        // all 128 cols -> butterfly over the low 4 lane bits
        float ss[8];
#pragma unroll
        for (int r = 0; r < 8; ++r) {
            float s = 0.f;
#pragma unroll
            for (int nt = 0; nt < 8; ++nt) s += acc[nt][r] * acc[nt][r];
            ss[r] = s;
        }
#pragma unroll
        for (int mk = 1; mk <= 8; mk <<= 1) {
#pragma unroll
            for (int r = 0; r < 8; ++r) ss[r] += __shfl_xor(ss[r], mk);
        }
        float scale[8];
#pragma unroll
        for (int r = 0; r < 8; ++r) scale[r] = 1.f / (sqrtf(ss[r]) + 1e-8f);
        unsigned int fl[8];
#pragma unroll
        for (int r = 0; r < 8; ++r) fl[r] = flags[tile * 16 + r + 8 * hh];
#pragma unroll
        for (int nt = 0; nt < 8; ++nt) {
#pragma unroll
            for (int r = 0; r < 8; ++r) {
                size_t g = (size_t)(tile * 16 + r + 8 * hh) * HID + nt * 16 + nl;
                hOut[g] = fl[r] ? acc[nt][r] * scale[r] : hIn[g];
            }
        }
    }
}

// ---------------- edge scatter: msg[dst] += h[src] ----------------
__global__ __launch_bounds__(256)
void msg_kernel(const float* __restrict__ h, const int* __restrict__ src,
                const int* __restrict__ dst, float* __restrict__ msg, int nE) {
    int e = blockIdx.x * 8 + (threadIdx.x >> 5);
    if (e >= nE) return;
    int lane = threadIdx.x & 31;
    int s = src[e], d = dst[e];
    float4 v = *((const float4*)(h + (size_t)s * HID) + lane);
    float* mp = msg + (size_t)d * HID + lane * 4;
    unsafeAtomicAdd(mp + 0, v.x);
    unsafeAtomicAdd(mp + 1, v.y);
    unsafeAtomicAdd(mp + 2, v.z);
    unsafeAtomicAdd(mp + 3, v.w);
}

__global__ void flags_kernel(const int* __restrict__ dst,
                             unsigned int* __restrict__ flags, int nE) {
    int e = blockIdx.x * blockDim.x + threadIdx.x;
    if (e < nE) flags[dst[e]] = 1u;
}

// ---------------- column sums + graph-vector normalize ----------------
__global__ void colsum_kernel(const float* __restrict__ h, float* __restrict__ gsum,
                              int nRows) {
    int c  = threadIdx.x;                 // 128 threads
    int r0 = blockIdx.x * 512;
    int r1 = r0 + 512; if (r1 > nRows) r1 = nRows;
    float s = 0.f;
    for (int r = r0; r < r1; ++r) s += h[(size_t)r * HID + c];
    unsafeAtomicAdd(&gsum[c], s);
}

__global__ void grnorm_kernel(const float* __restrict__ gsum, float* __restrict__ gr) {
    __shared__ float red[HID];
    int t = threadIdx.x;
    float v = gsum[t];
    red[t] = v * v;
    __syncthreads();
    for (int s = 64; s > 0; s >>= 1) {
        if (t < s) red[t] += red[t + s];
        __syncthreads();
    }
    gr[t] = v / (sqrtf(red[0]) + 1e-8f);
}

extern "C" void kernel_launch(void* const* d_in, const int* in_sizes, int n_in,
                              void* d_out, int out_size, void* d_ws, size_t ws_size,
                              hipStream_t stream) {
    const float* feat = (const float*)d_in[0];
    const int*   src  = (const int*)d_in[1];
    const int*   dst  = (const int*)d_in[2];
    const float* iW0 = (const float*)d_in[3];  const float* ib0 = (const float*)d_in[4];
    const float* iW1 = (const float*)d_in[5];  const float* ib1 = (const float*)d_in[6];
    const float* iW2 = (const float*)d_in[7];  const float* ib2 = (const float*)d_in[8];
    const float* nW0 = (const float*)d_in[9];  const float* nb0 = (const float*)d_in[10];
    const float* nW1 = (const float*)d_in[11]; const float* nb1 = (const float*)d_in[12];
    const float* nW2 = (const float*)d_in[13]; const float* nb2 = (const float*)d_in[14];

    const int nNodes = in_sizes[0] / 16;          // feat is [N,16]
    const int nE     = in_sizes[1];
    const int nTiles = (nNodes + 15) / 16;        // 3125 (exact for N=50000)
    const size_t HSZ = (size_t)nNodes * HID * sizeof(float);

    char* ws = (char*)d_ws;
    float*        hA    = (float*)ws;
    float*        hB    = (float*)(ws + HSZ);
    float*        msg   = (float*)(ws + 2 * HSZ);
    float*        gsum  = (float*)(ws + 3 * HSZ);
    float*        gr    = (float*)(ws + 3 * HSZ + 1024);
    unsigned int* flags = (unsigned int*)(ws + 3 * HSZ + 2048);

    // in-degree > 0 mask (iteration invariant)
    hipMemsetAsync(flags, 0, (size_t)nNodes * 4, stream);
    flags_kernel<<<(nE + 255) / 256, 256, 0, stream>>>(dst, flags, nE);

    const int blocks = (nTiles + WPB - 1) / WPB;
    init_mlp_kernel<<<blocks, WPB * 32, 0, stream>>>(feat, iW0, ib0, iW1, ib1,
                                                     iW2, ib2, hA, nTiles);
    hipMemsetAsync(gsum, 0, HID * sizeof(float), stream);
    colsum_kernel<<<(nNodes + 511) / 512, HID, 0, stream>>>(hA, gsum, nNodes);
    grnorm_kernel<<<1, HID, 0, stream>>>(gsum, gr);

    const float* hIn = hA;
    for (int it = 0; it < 3; ++it) {
        float* hOut = (it == 2) ? (float*)d_out : ((it == 0) ? hB : hA);
        hipMemsetAsync(msg, 0, HSZ, stream);
        msg_kernel<<<(nE + 7) / 8, 256, 0, stream>>>(hIn, src, dst, msg, nE);
        node_mlp_kernel<<<blocks, WPB * 32, 0, stream>>>(
            msg, hIn, gr, flags,
            nW0 + (size_t)it * K0 * HID,  nb0 + (size_t)it * HID,
            nW1 + (size_t)it * HID * HID, nb1 + (size_t)it * HID,
            nW2 + (size_t)it * HID * HID, nb2 + (size_t)it * HID,
            hOut, nTiles);
        if (it < 2) {
            hipMemsetAsync(gsum, 0, HID * sizeof(float), stream);
            colsum_kernel<<<(nNodes + 511) / 512, HID, 0, stream>>>(hOut, gsum, nNodes);
            grnorm_kernel<<<1, HID, 0, stream>>>(gsum, gr);
        }
        hIn = hOut;
    }
}
